// BCQ_34969623724656
// MI455X (gfx1250) — matrix operand order repaired and checked by reading, not compiled, not run on hardware
//
#include <hip/hip_runtime.h>
#include <hip/hip_bf16.h>
#include <math.h>

typedef _Float16 v2h  __attribute__((ext_vector_type(2)));
typedef _Float16 v4h  __attribute__((ext_vector_type(4)));
typedef _Float16 v16h __attribute__((ext_vector_type(16)));
typedef float    v8f  __attribute__((ext_vector_type(8)));
typedef float    f4   __attribute__((ext_vector_type(4)));
typedef float    f2   __attribute__((ext_vector_type(2)));

#define D_IN   292
#define D_PAD  320           // 10 K-tiles of 32
#define HID    256           // 8 K-tiles of 32
#define R_SZ   10
#define EPS    1e-5f
#define MB     128           // rows per block
#define NTHR   128           // 4 waves; each wave owns 2 M-tiles (32 rows)

// ---------------- workspace layout (f16 weights, K-padded) ----------------
#define W1_ELEMS (256*320)
#define W2_ELEMS (256*256)
#define OFF_W1V  0
#define OFF_W2V  (W1_ELEMS)
#define OFF_W1P  (W1_ELEMS + W2_ELEMS)
#define OFF_W2P  (2*W1_ELEMS + W2_ELEMS)
#define WTOTAL   (2*(W1_ELEMS + W2_ELEMS))

struct Params {
    const float *cum, *act, *pos;
    const float *n1g_v,*n1b_v,*b1_v,*n2g_v,*n2b_v,*b2_v,*n3g_v,*n3b_v,*w3_v,*b3_v;
    const float *n1g_p,*n1b_p,*b1_p,*n2g_p,*n2b_p,*b2_p,*n3g_p,*n3b_p,*w3_p,*b3_p;
    const _Float16 *W1h_v,*W2h_v,*W1h_p,*W2h_p;
    float* out;
};

__global__ void prep_weights(const float* __restrict__ w1v, const float* __restrict__ w2v,
                             const float* __restrict__ w1p, const float* __restrict__ w2p,
                             _Float16* __restrict__ ws) {
    for (int idx = blockIdx.x * blockDim.x + threadIdx.x; idx < WTOTAL;
         idx += gridDim.x * blockDim.x) {
        int i = idx;
        const float* src; _Float16* dst; int kpad, kcols;
        if (i < W1_ELEMS)                        { src = w1v; dst = ws + OFF_W1V; kpad = 320; kcols = 292; }
        else if ((i -= W1_ELEMS) < W2_ELEMS)     { src = w2v; dst = ws + OFF_W2V; kpad = 256; kcols = 256; }
        else if ((i -= W2_ELEMS) < W1_ELEMS)     { src = w1p; dst = ws + OFF_W1P; kpad = 320; kcols = 292; }
        else { i -= W1_ELEMS;                      src = w2p; dst = ws + OFF_W2P; kpad = 256; kcols = 256; }
        int n = i / kpad, k = i - n * kpad;
        float v = (k < kcols) ? src[n * kcols + k] : 0.0f;
        dst[i] = (_Float16)v;
    }
}

// ---- async global -> LDS staging of a weight chunk (CDNA5 ASYNCcnt path) ----
__device__ __forceinline__ void stage_async(const _Float16* gbase, unsigned ldsBase,
                                            int bytes, int tid) {
    unsigned long long g = (unsigned long long)(uintptr_t)gbase;
    for (int off = tid * 16; off < bytes; off += NTHR * 16) {
        unsigned l  = ldsBase + (unsigned)off;
        unsigned vo = (unsigned)off;
        asm volatile("global_load_async_to_lds_b128 %0, %1, %2"
                     :: "v"(l), "v"(vo), "s"(g) : "memory");
    }
}
__device__ __forceinline__ void wait_async() {
    asm volatile("s_wait_asynccnt 0x0" ::: "memory");
}

// ---- A fragment 16x32 f16 (ISA lane striping), optional LN-affine via f16 tables ----
template<int KPAD, bool AFFINE>
__device__ __forceinline__ v16h load_a_frag(const _Float16* __restrict__ A,
                                            const _Float16* __restrict__ gt,
                                            const _Float16* __restrict__ bt,
                                            int mrow, int lane, int kt) {
    v16h a;
    const int row   = mrow + (lane & 15);
    const int khalf = (lane & 16) ? 8 : 0;
    const _Float16* Ar = A + row * KPAD + kt * 32;
    const _Float16* gr = gt + kt * 32;
    const _Float16* br = bt + kt * 32;
#pragma unroll
    for (int v = 0; v < 8; ++v) {
        int kb = ((v < 4) ? 0 : 16) + khalf + 2 * (v & 3);
        _Float16 x0 = Ar[kb], x1 = Ar[kb + 1];
        if (AFFINE) {                      // packed f16 FMA expected
            x0 = x0 * gr[kb]     + br[kb];
            x1 = x1 * gr[kb + 1] + br[kb + 1];
        }
        a[2 * v] = x0; a[2 * v + 1] = x1;
    }
    return a;
}

// ---- B fragment 32x16 f16 from LDS-staged chunk (64 W-rows x KPAD, row-major) ----
template<int KPAD>
__device__ __forceinline__ v16h load_b_frag(const _Float16* __restrict__ Wc, int ntl,
                                            int lane, int kt) {
    v16h b;
    const int nl = ntl * 16 + (lane & 15);
    const _Float16* Wr = Wc + nl * KPAD + kt * 32 + ((lane & 16) ? 8 : 0);
#pragma unroll
    for (int v = 0; v < 8; ++v) { int kb = ((v < 4) ? 0 : 16) + 2 * (v & 3); b[2*v] = Wr[kb]; b[2*v+1] = Wr[kb+1]; }
    return b;
}

// One dense layer, block-cooperative: H[128][256] = relu(A @ W^T + bias), f16 out.
// Weights double-buffered in LDS via async copies; every wave does 2 M-tiles x 4 N-tiles.
template<int KT, int KPAD, bool AFFINE>
__device__ void block_layer(const _Float16* __restrict__ A,
                            const _Float16* __restrict__ gt, const _Float16* __restrict__ bt,
                            const _Float16* __restrict__ Wg,
                            const _Float16* wb0, const _Float16* wb1,
                            unsigned wb0Lds, unsigned wb1Lds,
                            const float* __restrict__ bias, _Float16* __restrict__ Hout,
                            int tid, int lane, int wave) {
    const int CHUNK_BYTES = 128 * KPAD;           // 64 rows * KPAD halfs * 2B
    stage_async(Wg, wb0Lds, CHUNK_BYTES, tid);    // prefetch chunk 0
    wait_async();
    __syncthreads();
    const int mbase = wave * 32;
    for (int ntg = 0; ntg < 4; ++ntg) {
        const _Float16* cur = (ntg & 1) ? wb1 : wb0;
        if (ntg < 3)   // prefetch next chunk while computing on current
            stage_async(Wg + (ntg + 1) * 64 * KPAD, (ntg & 1) ? wb0Lds : wb1Lds,
                        CHUNK_BYTES, tid);
        v8f acc[2][4] = {};
        for (int kt = 0; kt < KT; ++kt) {
            v16h bf[4];
#pragma unroll
            for (int j = 0; j < 4; ++j) bf[j] = load_b_frag<KPAD>(cur, j, lane, kt);
#pragma unroll
            for (int m = 0; m < 2; ++m) {
                v16h a = load_a_frag<KPAD, AFFINE>(A, gt, bt, mbase + m * 16, lane, kt);
#pragma unroll
                for (int j = 0; j < 4; ++j)
                    acc[m][j] = __builtin_amdgcn_wmma_f32_16x16x32_f16(
                        false, a, false, bf[j], (short)0, acc[m][j], false, false);
            }
        }
        const int n = lane & 15;
#pragma unroll
        for (int m = 0; m < 2; ++m) {
            const int rbase = mbase + m * 16 + ((lane & 16) ? 8 : 0);
#pragma unroll
            for (int j = 0; j < 4; ++j) {
                const int col = (ntg * 4 + j) * 16 + n;
                const float bv = bias[col];
#pragma unroll
                for (int e = 0; e < 8; ++e) {
                    float val = acc[m][j][e] + bv;
                    val = val > 0.0f ? val : 0.0f;
                    Hout[(rbase + e) * HID + col] = (_Float16)val;
                }
            }
        }
        wait_async();
        __syncthreads();
    }
}

// per-row LayerNorm stats over f16 data (thread r owns row r)
__device__ __forceinline__ void row_stats(const _Float16* __restrict__ X, int stride,
                                          int nquads, float inv_n,
                                          float* __restrict__ mean, float* __restrict__ rstd,
                                          int tid) {
    const v4h* xr = (const v4h*)(X + tid * stride);
    float s = 0.0f;
    for (int q = 0; q < nquads; ++q) {
        v4h x = xr[q];
        s += (float)x[0] + (float)x[1] + (float)x[2] + (float)x[3];
    }
    float m = s * inv_n, var = 0.0f;
    for (int q = 0; q < nquads; ++q) {
        v4h x = xr[q];
#pragma unroll
        for (int e = 0; e < 4; ++e) { float d = (float)x[e] - m; var += d * d; }
    }
    mean[tid] = m;
    rstd[tid] = rsqrtf(var * inv_n + EPS);
}

__global__ __launch_bounds__(NTHR) void bcq_fused(Params p, int nrows) {
    extern __shared__ char smem[];
    _Float16* xhat = (_Float16*)smem;                     // [128][320] f16
    _Float16* h1   = xhat + MB * D_PAD;                   // [128][256] f16
    _Float16* h2   = h1 + MB * HID;                       // [128][256] f16
    _Float16* wb0  = h2 + MB * HID;                       // 40 KB weight stage
    _Float16* wb1  = wb0 + 64 * D_PAD;                    // 40 KB weight stage
    _Float16* gtab = wb1 + 64 * D_PAD;                    // [320] f16
    _Float16* btab = gtab + D_PAD;                        // [320] f16
    float* meanx = (float*)(btab + D_PAD);                // [128]
    float* rstdx = meanx + MB;
    float* meanh = rstdx + MB;
    float* rstdh = meanh + MB;
    float* sc    = rstdh + MB;                            // [128][10]

    const int tid  = threadIdx.x;
    const int lane = tid & 31;
    const int wave = tid >> 5;
    const long grow0 = (long)blockIdx.x * MB;

    const unsigned ldsStatic = (unsigned)__builtin_amdgcn_groupstaticsize();
    const unsigned wb0Lds = ldsStatic + (unsigned)((char*)wb0 - (char*)smem);
    const unsigned wb1Lds = ldsStatic + (unsigned)((char*)wb1 - (char*)smem);

    // --- stage 0: gather concat([cum|act|pos]) -> f16 LDS (aligned float2 loads) ---
    for (int u = tid; u < MB * 160; u += NTHR) {          // 160 half-pairs per row
        int r = u / 160, pr = u - r * 160;
        long gr = grow0 + r;
        f2 xv = {0.0f, 0.0f};
        if (gr < nrows) {
            if (pr < 73)        xv = *(const f2*)(p.cum + gr * 146 + pr * 2);
            else if (pr < 141)  xv = *(const f2*)(p.act + gr * 136 + (pr - 73) * 2);
            else if (pr < 146)  xv = *(const f2*)(p.pos + gr * 10 + (pr - 141) * 2);
        }
        v2h o; o[0] = (_Float16)xv[0]; o[1] = (_Float16)xv[1];
        *(v2h*)(xhat + r * D_PAD + pr * 2) = o;
    }
    __syncthreads();
    row_stats(xhat, D_PAD, 73, 1.0f / (float)D_IN, meanx, rstdx, tid);
    __syncthreads();
    // in-place x -> (x-mean)*rstd (shared by both towers; affine folded into A-frag build)
    for (int u = tid; u < MB * 80; u += NTHR) {
        int r = u / 80, q = u - r * 80;
        v4h* ptr = (v4h*)(xhat + r * D_PAD + q * 4);
        if (q < 73) {
            _Float16 m = (_Float16)meanx[r], rs = (_Float16)rstdx[r];
            v4h mm = {m, m, m, m}, rr = {rs, rs, rs, rs};
            *ptr = (*ptr - mm) * rr;                      // packed f16 math
        } else {
            *ptr = v4h{(_Float16)0, (_Float16)0, (_Float16)0, (_Float16)0};
        }
    }
    __syncthreads();

    for (int t = 0; t < 2; ++t) {
        const float *n1g = t ? p.n1g_p : p.n1g_v, *n1b = t ? p.n1b_p : p.n1b_v;
        const float *b1  = t ? p.b1_p  : p.b1_v;
        const float *n2g = t ? p.n2g_p : p.n2g_v, *n2b = t ? p.n2b_p : p.n2b_v;
        const float *b2  = t ? p.b2_p  : p.b2_v;
        const float *n3g = t ? p.n3g_p : p.n3g_v, *n3b = t ? p.n3b_p : p.n3b_v;
        const _Float16 *W1h = t ? p.W1h_p : p.W1h_v;
        const _Float16 *W2h = t ? p.W2h_p : p.W2h_v;

        // LN1 affine tables (f16, zero K-pad)
        for (int i = tid; i < D_PAD; i += NTHR) {
            gtab[i] = (_Float16)(i < D_IN ? n1g[i] : 0.0f);
            btab[i] = (_Float16)(i < D_IN ? n1b[i] : 0.0f);
        }
        __syncthreads();

        // layer 1: 292->256 (WMMA, async-staged weights)
        block_layer<10, D_PAD, true>(xhat, gtab, btab, W1h, wb0, wb1, wb0Lds, wb1Lds,
                                     b1, h1, tid, lane, wave);
        // LN2 (in-place, with affine)
        row_stats(h1, HID, 64, 1.0f / (float)HID, meanh, rstdh, tid);
        __syncthreads();
        for (int u = tid; u < MB * 64; u += NTHR) {
            int r = u / 64, q = u - r * 64, k = q * 4;
            v4h* ptr = (v4h*)(h1 + r * HID + k);
            v4h h = *ptr;
            f4 g = *(const f4*)(n2g + k), b = *(const f4*)(n2b + k);
            float m = meanh[r], rs = rstdh[r];
            v4h o;
#pragma unroll
            for (int e = 0; e < 4; ++e) o[e] = (_Float16)((((float)h[e]) - m) * rs * g[e] + b[e]);
            *ptr = o;
        }
        __syncthreads();

        // layer 2: 256->256
        block_layer<8, HID, false>(h1, nullptr, nullptr, W2h, wb0, wb1, wb0Lds, wb1Lds,
                                   b2, h2, tid, lane, wave);
        // LN3 (in-place, with affine)
        row_stats(h2, HID, 64, 1.0f / (float)HID, meanh, rstdh, tid);
        __syncthreads();
        for (int u = tid; u < MB * 64; u += NTHR) {
            int r = u / 64, q = u - r * 64, k = q * 4;
            v4h* ptr = (v4h*)(h2 + r * HID + k);
            v4h h = *ptr;
            f4 g = *(const f4*)(n3g + k), b = *(const f4*)(n3b + k);
            float m = meanh[r], rs = rstdh[r];
            v4h o;
#pragma unroll
            for (int e = 0; e < 4; ++e) o[e] = (_Float16)((((float)h[e]) - m) * rs * g[e] + b[e]);
            *ptr = o;
        }
        __syncthreads();

        if (t == 0) {
            // value head 256->1, one row per thread
            long gr = grow0 + tid;
            if (gr < nrows) {
                float s = p.b3_v[0];
                const v4h* hr = (const v4h*)(h2 + tid * HID);
                for (int q = 0; q < 64; ++q) {
                    v4h h = hr[q];
                    f4 w = *(const f4*)(p.w3_v + q * 4);
#pragma unroll
                    for (int e = 0; e < 4; ++e) s += (float)h[e] * w[e];
                }
                p.out[gr] = s;
            }
        } else {
            // prob head 256->10
            for (int u = tid; u < MB * R_SZ; u += NTHR) {
                int r = u / R_SZ, j = u - r * R_SZ;
                float s = p.b3_p[j];
                const v4h* hr = (const v4h*)(h2 + r * HID);
                for (int q = 0; q < 64; ++q) {
                    v4h h = hr[q];
                    f4 w = *(const f4*)(p.w3_p + j * HID + q * 4);
#pragma unroll
                    for (int e = 0; e < 4; ++e) s += (float)h[e] * w[e];
                }
                sc[u] = s;
            }
            __syncthreads();
            long gr = grow0 + tid;
            if (gr < nrows) {
                float mx = -INFINITY;
                for (int j = 0; j < R_SZ; ++j) mx = fmaxf(mx, sc[tid * R_SZ + j]);
                float se = 0.0f;
                for (int j = 0; j < R_SZ; ++j) se += expf(sc[tid * R_SZ + j] - mx);
                float lse = mx + logf(se);
                for (int j = 0; j < R_SZ; ++j) {
                    float v = sc[tid * R_SZ + j];
                    p.out[(long)nrows + gr * R_SZ + j]      = v - lse;  // log_prob
                    p.out[(long)nrows * 11 + gr * R_SZ + j] = v;        // score
                }
            }
        }
        __syncthreads();
    }
}

extern "C" void kernel_launch(void* const* d_in, const int* in_sizes, int n_in,
                              void* d_out, int out_size, void* d_ws, size_t ws_size,
                              hipStream_t stream) {
    _Float16* wsH = (_Float16*)d_ws;

    prep_weights<<<576, 256, 0, stream>>>((const float*)d_in[3],  (const float*)d_in[5],
                                          (const float*)d_in[15], (const float*)d_in[17], wsH);

    int nrows = in_sizes[2] / R_SZ;

    Params p;
    p.cum = (const float*)d_in[0];
    p.act = (const float*)d_in[1];
    p.pos = (const float*)d_in[2];
    p.n1g_v = (const float*)d_in[9];  p.n1b_v = (const float*)d_in[10];
    p.b1_v  = (const float*)d_in[4];
    p.n2g_v = (const float*)d_in[11]; p.n2b_v = (const float*)d_in[12];
    p.b2_v  = (const float*)d_in[6];
    p.n3g_v = (const float*)d_in[13]; p.n3b_v = (const float*)d_in[14];
    p.w3_v  = (const float*)d_in[7];  p.b3_v  = (const float*)d_in[8];
    p.n1g_p = (const float*)d_in[21]; p.n1b_p = (const float*)d_in[22];
    p.b1_p  = (const float*)d_in[16];
    p.n2g_p = (const float*)d_in[23]; p.n2b_p = (const float*)d_in[24];
    p.b2_p  = (const float*)d_in[18];
    p.n3g_p = (const float*)d_in[25]; p.n3b_p = (const float*)d_in[26];
    p.w3_p  = (const float*)d_in[19]; p.b3_p  = (const float*)d_in[20];
    p.W1h_v = wsH + OFF_W1V; p.W2h_v = wsH + OFF_W2V;
    p.W1h_p = wsH + OFF_W1P; p.W2h_p = wsH + OFF_W2P;
    p.out   = (float*)d_out;

    size_t smem = (size_t)(MB * D_PAD + 2 * MB * HID + 2 * 64 * D_PAD + 2 * D_PAD) * sizeof(_Float16)
                + (size_t)(4 * MB + MB * R_SZ) * sizeof(float);

    int nblocks = (nrows + MB - 1) / MB;
    bcq_fused<<<nblocks, NTHR, smem, stream>>>(p, nrows);
}
